// MAB_82643760709741
// MI455X (gfx1250) — matrix operand order repaired and checked by reading, not compiled, not run on hardware
//
#include <hip/hip_runtime.h>
#include <hip/hip_bf16.h>

// ---------------- types / constants ----------------
typedef __bf16 bf16_t;
typedef __attribute__((ext_vector_type(16))) __bf16 v16bf;
typedef __attribute__((ext_vector_type(8)))  __bf16 v8bf;
typedef __attribute__((ext_vector_type(8)))  float  v8f;

union Frag { v16bf v; v8bf h[2]; };

constexpr int kB = 4, kSQ = 2048, kSK = 2048, kD = 512, kH = 8, kDH = 64;

__device__ __forceinline__ v8f wmma_bf16(v16bf a, v16bf b, v8f c) {
  // D = A(16x32 bf16) * B(32x16 bf16) + C(16x16 f32)
  return __builtin_amdgcn_wmma_f32_16x16x32_bf16(false, a, false, b, (short)0, c,
                                                 false, false);
}

__device__ __forceinline__ v8f v8f_zero() {
  v8f z;
#pragma unroll
  for (int i = 0; i < 8; ++i) z[i] = 0.0f;
  return z;
}

// LDS byte offset of a generic pointer (flat LDS addr: low 32 bits = offset)
__device__ __forceinline__ unsigned lds_off(const void* p) {
  return (unsigned)(size_t)p;
}

// async 16B global -> LDS copy (per lane), tracked with ASYNCcnt
__device__ __forceinline__ void async_cp16(unsigned lds_byte, unsigned long long gaddr) {
  asm volatile("global_load_async_to_lds_b128 %0, %1, off"
               :: "v"(lds_byte), "v"(gaddr) : "memory");
}

// ---------------- f32 -> bf16 convert ----------------
__global__ __launch_bounds__(256) void cvt_kernel(const float* __restrict__ x,
                                                  bf16_t* __restrict__ y, int n) {
  int i = blockIdx.x * 256 + threadIdx.x;
  if (i < n) y[i] = (bf16_t)x[i];
}

// Wt[n*D + k] = W[k*D + n]   (square D x D), converted to bf16
__global__ __launch_bounds__(256) void wtrans_kernel(const float* __restrict__ W,
                                                     bf16_t* __restrict__ Wt) {
  int i = blockIdx.x * 256 + threadIdx.x;  // over D*D
  int n = i / kD, k = i % kD;
  Wt[i] = (bf16_t)W[(size_t)k * kD + n];
}

// ---------------- WMMA GEMM:  C[M,D] = A[M,D] @ W + bias (+resid)(relu) --------
// A row-major bf16 (streamed from global); Wt = W^T row-major bf16 ([N][K]).
// The block's 128x512 weight tile (contiguous 128KB of Wt) is DMA'd to LDS once
// via global_load_async_to_lds_b128 with an XOR granule swizzle (bank-conflict
// free B-fragment ds_load_b128 reads), then all 8 waves share it.
// Wave tile = 16 rows x 128 cols: 8 WMMAs per A fragment -> A re-read only 4x.
enum { OUT_BF = 0, OUT_VT = 1, OUT_F32 = 2 };

template <int OMODE, bool RELU, bool RESID>
__global__ __launch_bounds__(256) void gemm_kernel(
    const bf16_t* __restrict__ Abf, const bf16_t* __restrict__ Wt,
    const float* __restrict__ bias, const float* __restrict__ resid,
    void* __restrict__ out) {
  __shared__ __align__(16) bf16_t Wlds[128 * kD];  // 128KB, granule-swizzled
  const int lane = threadIdx.x & 31;
  const int wave = threadIdx.x >> 5;
  const int hf   = lane >> 4;
  const int l15  = lane & 15;
  const int rowBase = blockIdx.y * 128 + wave * 16;
  const int colBase = blockIdx.x * 128;

  // --- stage weight tile: 128 rows x 1KB, contiguous 128KB at Wt+colBase*kD --
  {
    const unsigned base = lds_off(&Wlds[0]);
    const unsigned long long gsrc =
        (unsigned long long)(Wt + (size_t)colBase * kD);
#pragma unroll
    for (int it = 0; it < 32; ++it) {
      const unsigned off = (threadIdx.x + it * 256) * 16;  // byte offset
      const unsigned row = off >> 10;                      // 1KB rows
      const unsigned g   = (off >> 4) & 63;                // 16B granule in row
      const unsigned lo  = base + (row << 10) + (((g ^ row) & 63) << 4);
      async_cp16(lo, gsrc + off);
    }
    asm volatile("s_wait_asynccnt 0x0" ::: "memory");
    __syncthreads();
  }

  const bf16_t* arow = Abf + (size_t)(rowBase + l15) * kD;
  v8f acc[8];
#pragma unroll
  for (int t = 0; t < 8; ++t) acc[t] = v8f_zero();

  for (int kk = 0; kk < kD; kk += 32) {
    Frag af;  // A 16x32: lane row=l15; K = hf*16+{0..15}
    af.h[0] = *(const v8bf*)(arow + kk + hf * 16);
    af.h[1] = *(const v8bf*)(arow + kk + hf * 16 + 8);
    const unsigned g0 = (unsigned)(kk >> 3) + hf * 2;  // 16B granule index
#pragma unroll
    for (int t = 0; t < 8; ++t) {
      const int nl = t * 16 + l15;  // local column (0..127)
      const char* wb = (const char*)Wlds + (nl << 10);
      Frag wf;
      wf.h[0] = *(const v8bf*)(wb + ((((g0 + 0) ^ nl) & 63) << 4));
      wf.h[1] = *(const v8bf*)(wb + ((((g0 + 1) ^ nl) & 63) << 4));
      acc[t] = wmma_bf16(af.v, wf.v, acc[t]);
    }
  }

#pragma unroll
  for (int t = 0; t < 8; ++t) {
    const int n = colBase + t * 16 + l15;
    const float bb = bias[n];
#pragma unroll
    for (int j = 0; j < 8; ++j) {
      const int row = rowBase + hf * 8 + j;
      float x = acc[t][j] + bb;
      if (RESID) x += resid[(size_t)row * kD + n];
      if (RELU)  x = fmaxf(x, 0.0f);
      if (OMODE == OUT_BF) {
        ((bf16_t*)out)[(size_t)row * kD + n] = (bf16_t)x;
      } else if (OMODE == OUT_VT) {  // v transposed: [(b*D + n)][s]
        const int b = row / kSK, s = row % kSK;
        ((bf16_t*)out)[((size_t)(b * kD + n)) * kSK + s] = (bf16_t)x;
      } else {
        ((float*)out)[(size_t)row * kD + n] = x;
      }
    }
  }
}

// ---------------- flash attention (transposed formulation) ----------------
// grid = (SQ/128, B*H); each wave owns 16 q rows x DH=64.
// Computes S^T = K_tile x q^T and ctx^T = V^T x P^T so that softmax row stats
// are IN-LANE reductions (lane column = query row): no per-row bpermute storms.
// K/V 32-key tiles are double-buffered in LDS via async DMA, shared by 8 waves.
__global__ __launch_bounds__(256) void attn_kernel(
    const bf16_t* __restrict__ qbf,  // [B*SQ, D]
    const bf16_t* __restrict__ kbf,  // [B*SK, D]
    const bf16_t* __restrict__ vT,   // [(b*D + h*DH+dh)][SK]
    const unsigned char* __restrict__ qmask,  // [B,SQ]
    const unsigned char* __restrict__ kmask,  // [B,SK]
    bf16_t* __restrict__ ctx) {               // [B*SQ, D]
  __shared__ __align__(16) bf16_t Klds[2][32][72];  // 32 keys x 64 dh (+8 pad)
  __shared__ __align__(16) bf16_t Vlds[2][64][40];  // 64 dh x 32 keys (+8 pad)
  __shared__ __align__(16) bf16_t Tbuf[8][16][40];  // P^T: [qrow][key] (+8 pad)
  const int tid  = threadIdx.x;
  const int lane = tid & 31;
  const int wave = tid >> 5;
  const int hf   = lane >> 4;
  const int l15  = lane & 15;
  const int bh = blockIdx.y;
  const int b = bh / kH, h = bh % kH;
  const int q0 = blockIdx.x * 128 + wave * 16;
  const float scale = 0.044194173824159216f;  // 1/sqrt(D), D=512 (reference!)

  // staging thread mapping: K tile = 32 rows x 128B; V tile = 64 rows x 64B
  const int krow_s = tid >> 3, kcol_s = tid & 7;  // 256 thr x 16B = 4KB
  const int vrow_s = tid >> 2, vcol_s = tid & 3;  // 256 thr x 16B = 4KB
  const size_t kbase = (size_t)(b * kSK) * kD + (size_t)h * kDH;
  const size_t vbase = (size_t)(b * kD + h * kDH) * kSK;

  // q B-fragments (col = qrow = l15; K = dh, 16 contiguous per half-wave)
  const bf16_t* qrp = qbf + (size_t)(b * kSQ + q0 + l15) * kD + h * kDH;
  Frag q0f, q1f;
  q0f.h[0] = *(const v8bf*)(qrp + hf * 16);
  q0f.h[1] = *(const v8bf*)(qrp + hf * 16 + 8);
  q1f.h[0] = *(const v8bf*)(qrp + 32 + hf * 16);
  q1f.h[1] = *(const v8bf*)(qrp + 32 + hf * 16 + 8);
  const bool qmA = qmask[b * kSQ + q0 + l15] != 0;  // per-lane (per q row)

  v8f acc[4];  // ctx^T: acc[t] rows dh = t*16+hf*8+j, col qrow = l15
#pragma unroll
  for (int t = 0; t < 4; ++t) acc[t] = v8f_zero();
  float mA = -3.0e30f, lA = 0.0f;  // running max / denom for this lane's q row

  // issue DMA for the first K/V stage
  async_cp16(lds_off(&Klds[0][krow_s][0]) + kcol_s * 16,
             (unsigned long long)(kbf + kbase + (size_t)krow_s * kD) + kcol_s * 16);
  async_cp16(lds_off(&Vlds[0][vrow_s][0]) + vcol_s * 16,
             (unsigned long long)(vT + vbase + (size_t)vrow_s * kSK) + vcol_s * 16);

  for (int k0 = 0; k0 < kSK; k0 += 32) {
    const int cur = (k0 >> 5) & 1;
    const bool more = (k0 + 32) < kSK;
    if (more) {  // prefetch next stage into the other buffer
      const int nk0 = k0 + 32;
      async_cp16(lds_off(&Klds[cur ^ 1][krow_s][0]) + kcol_s * 16,
                 (unsigned long long)(kbf + kbase + (size_t)(nk0 + krow_s) * kD) +
                     kcol_s * 16);
      async_cp16(lds_off(&Vlds[cur ^ 1][vrow_s][0]) + vcol_s * 16,
                 (unsigned long long)(vT + vbase + (size_t)vrow_s * kSK + nk0) +
                     vcol_s * 16);
      asm volatile("s_wait_asynccnt 0x2" ::: "memory");  // stage `cur` landed
    } else {
      asm volatile("s_wait_asynccnt 0x0" ::: "memory");
    }
    __syncthreads();

    // S^T tiles: A = K rows (16 keys x 64 dh), B = q^T; C col = q row
    v8f st[2];
#pragma unroll
    for (int t = 0; t < 2; ++t) {
      const bf16_t* kr = &Klds[cur][t * 16 + l15][0];  // row = key
      Frag ka0, ka1;
      ka0.h[0] = *(const v8bf*)(kr + hf * 16);
      ka0.h[1] = *(const v8bf*)(kr + hf * 16 + 8);
      ka1.h[0] = *(const v8bf*)(kr + 32 + hf * 16);
      ka1.h[1] = *(const v8bf*)(kr + 32 + hf * 16 + 8);
      v8f s = v8f_zero();
      s = wmma_bf16(ka0.v, q0f.v, s);
      s = wmma_bf16(ka1.v, q1f.v, s);
      st[t] = s;
    }

    // scale + masks; element (t,j) is key = k0 + t*16 + hf*8 + j
    float c[2][8];
#pragma unroll
    for (int t = 0; t < 2; ++t) {
      const unsigned long long km8 =
          *(const unsigned long long*)(kmask + (size_t)b * kSK + k0 + t * 16 + hf * 8);
#pragma unroll
      for (int j = 0; j < 8; ++j) {
        const bool ok = qmA && (((km8 >> (8 * j)) & 1ull) != 0);
        c[t][j] = ok ? st[t][j] * scale : -1.0e6f;
      }
    }

    // in-lane row stats (lane column = q row); combine halves with one shuffle
    float mt = c[0][0];
#pragma unroll
    for (int t = 0; t < 2; ++t)
#pragma unroll
      for (int j = 0; j < 8; ++j) mt = fmaxf(mt, c[t][j]);
    mt = fmaxf(mt, __shfl_xor(mt, 16, 32));
    const float mnew = fmaxf(mA, mt);
    const float corr = __expf(mA - mnew);
    mA = mnew;
    float p[2][8];
    float ts = 0.0f;
#pragma unroll
    for (int t = 0; t < 2; ++t)
#pragma unroll
      for (int j = 0; j < 8; ++j) { p[t][j] = __expf(c[t][j] - mnew); ts += p[t][j]; }
    ts += __shfl_xor(ts, 16, 32);
    lA = lA * corr + ts;
#pragma unroll
    for (int t = 0; t < 4; ++t)
#pragma unroll
      for (int j = 0; j < 8; ++j) acc[t][j] *= corr;  // corr uniform per lane

    // write P^T (C layout) to LDS; re-read as B fragment [qrow][16 contig keys]
#pragma unroll
    for (int t = 0; t < 2; ++t)
#pragma unroll
      for (int j = 0; j < 8; ++j)
        Tbuf[wave][l15][t * 16 + hf * 8 + j] = (bf16_t)p[t][j];
    __asm__ __volatile__("" ::: "memory");  // keep ds_store -> ds_load ordered
    Frag pb;
    pb.h[0] = *(const v8bf*)(&Tbuf[wave][l15][hf * 16]);
    pb.h[1] = *(const v8bf*)(&Tbuf[wave][l15][hf * 16 + 8]);

    // ctx^T += V^T x P^T : A = V rows (16 dh x 32 keys), B = P^T
#pragma unroll
    for (int t = 0; t < 4; ++t) {
      const bf16_t* vr = &Vlds[cur][t * 16 + l15][0];  // row = dh
      Frag va;
      va.h[0] = *(const v8bf*)(vr + hf * 16);
      va.h[1] = *(const v8bf*)(vr + hf * 16 + 8);
      acc[t] = wmma_bf16(va.v, pb.v, acc[t]);
    }
    __syncthreads();  // all waves done with buffer `cur` before it is reused
  }

  // epilogue: lane = q row; 8 consecutive dh per accumulator -> b128 stores
  const float linv = 1.0f / lA;
  bf16_t* crow = ctx + (size_t)(b * kSQ + q0 + l15) * kD + h * kDH;
#pragma unroll
  for (int t = 0; t < 4; ++t) {
    v8bf o;
#pragma unroll
    for (int j = 0; j < 8; ++j) o[j] = (bf16_t)(acc[t][j] * linv);
    *(v8bf*)(crow + t * 16 + hf * 8) = o;
  }
}

// ---------------- LayerNorm (block per row, D=512, 128 threads) ----------------
template <bool ADD, bool WRITE_BF>
__global__ __launch_bounds__(128) void ln_kernel(
    const float* __restrict__ X, const float* __restrict__ Y,
    const float* __restrict__ g, const float* __restrict__ bta,
    float* __restrict__ outf, bf16_t* __restrict__ outbf) {
  const int row = blockIdx.x;
  const int tid = threadIdx.x;
  float v[4];
  float s = 0.0f, s2 = 0.0f;
#pragma unroll
  for (int i = 0; i < 4; ++i) {
    const int c = tid + i * 128;
    float x = X[(size_t)row * kD + c];
    if (ADD) x += Y[(size_t)row * kD + c];
    v[i] = x; s += x; s2 += x * x;
  }
#pragma unroll
  for (int off = 1; off < 32; off <<= 1) {
    s  += __shfl_xor(s, off, 32);
    s2 += __shfl_xor(s2, off, 32);
  }
  __shared__ float sm[4], sm2[4];
  const int wv = tid >> 5;
  if ((tid & 31) == 0) { sm[wv] = s; sm2[wv] = s2; }
  __syncthreads();
  s  = sm[0] + sm[1] + sm[2] + sm[3];
  s2 = sm2[0] + sm2[1] + sm2[2] + sm2[3];
  const float mean = s * (1.0f / kD);
  const float var  = s2 * (1.0f / kD) - mean * mean;
  const float inv  = rsqrtf(var + 1e-5f);
#pragma unroll
  for (int i = 0; i < 4; ++i) {
    const int c = tid + i * 128;
    const float o = (v[i] - mean) * inv * g[c] + bta[c];
    outf[(size_t)row * kD + c] = o;
    if (WRITE_BF) outbf[(size_t)row * kD + c] = (bf16_t)o;
  }
}

// ---------------- launcher ----------------
extern "C" void kernel_launch(void* const* d_in, const int* in_sizes, int n_in,
                              void* d_out, int out_size, void* d_ws, size_t ws_size,
                              hipStream_t stream) {
  const float* Q     = (const float*)d_in[0];
  const float* K     = (const float*)d_in[1];
  const unsigned char* qmask = (const unsigned char*)d_in[2];
  const unsigned char* kmask = (const unsigned char*)d_in[3];
  const float* Wq = (const float*)d_in[4];  const float* bq = (const float*)d_in[5];
  const float* Wk = (const float*)d_in[6];  const float* bk = (const float*)d_in[7];
  const float* Wv = (const float*)d_in[8];  const float* bv = (const float*)d_in[9];
  const float* Wo = (const float*)d_in[10]; const float* bo = (const float*)d_in[11];
  const float* ln0_g = (const float*)d_in[12]; const float* ln0_b = (const float*)d_in[13];
  const float* W1 = (const float*)d_in[14]; const float* b1 = (const float*)d_in[15];
  const float* W2 = (const float*)d_in[16]; const float* b2 = (const float*)d_in[17];
  const float* ln1_g = (const float*)d_in[18]; const float* ln1_b = (const float*)d_in[19];

  const size_t MD = (size_t)kB * kSQ * kD;  // 4M elements
  char* ws = (char*)d_ws;
  size_t off = 0;
  auto alloc = [&](size_t bytes) {
    size_t o = off; off += (bytes + 255) & ~(size_t)255; return ws + o;
  };

  bf16_t* Qbf = (bf16_t*)alloc(MD * 2);
  bf16_t* Kbf = (bf16_t*)alloc(MD * 2);
  bf16_t* WT[6];
  for (int i = 0; i < 6; ++i) WT[i] = (bf16_t*)alloc((size_t)kD * kD * 2);
  bf16_t* qb   = (bf16_t*)alloc(MD * 2);
  bf16_t* kb   = (bf16_t*)alloc(MD * 2);
  bf16_t* vt   = (bf16_t*)alloc(MD * 2);
  float*  Hpre = (float*)alloc(MD * 4);
  float*  Hln  = (float*)alloc(MD * 4);
  // aliases of dead buffers (footprint control)
  bf16_t* ctxb = Qbf;           // Qbf dead after q projection
  bf16_t* Hbf  = Kbf;           // Kbf dead after k/v projections
  bf16_t* M1   = qb;            // qb dead after attention
  float*  mlp  = Hpre;          // Hpre dead after ln0

  const dim3 blk256(256), blk128(128);
  const dim3 gGemm(kD / 128, (kB * kSQ) / 128);      // (4, 64)
  const dim3 gAttn(kSQ / 128, kB * kH);              // (16, 32)

  // 1) convert inputs + weights
  cvt_kernel<<<(int)(MD / 256), blk256, 0, stream>>>(Q, Qbf, (int)MD);
  cvt_kernel<<<(int)(MD / 256), blk256, 0, stream>>>(K, Kbf, (int)MD);
  const float* Ws[6] = {Wq, Wk, Wv, Wo, W1, W2};
  for (int i = 0; i < 6; ++i)
    wtrans_kernel<<<(kD * kD) / 256, blk256, 0, stream>>>(Ws[i], WT[i]);

  // 2) projections
  gemm_kernel<OUT_BF, false, false><<<gGemm, blk256, 0, stream>>>(Qbf, WT[0], bq, nullptr, qb);
  gemm_kernel<OUT_BF, false, false><<<gGemm, blk256, 0, stream>>>(Kbf, WT[1], bk, nullptr, kb);
  gemm_kernel<OUT_VT, false, false><<<gGemm, blk256, 0, stream>>>(Kbf, WT[2], bv, nullptr, vt);

  // 3) attention
  attn_kernel<<<gAttn, blk256, 0, stream>>>(qb, kb, vt, qmask, kmask, ctxb);

  // 4) output projection + residual, then LN0
  gemm_kernel<OUT_F32, false, true><<<gGemm, blk256, 0, stream>>>(ctxb, WT[3], bo, Q, Hpre);
  ln_kernel<false, true><<<kB * kSQ, blk128, 0, stream>>>(Hpre, nullptr, ln0_g, ln0_b, Hln, Hbf);

  // 5) MLP (relu on both layers), then residual + LN1 -> output
  gemm_kernel<OUT_BF, true, false><<<gGemm, blk256, 0, stream>>>(Hbf, WT[4], b1, nullptr, M1);
  gemm_kernel<OUT_F32, true, false><<<gGemm, blk256, 0, stream>>>(M1, WT[5], b2, nullptr, mlp);
  ln_kernel<true, false><<<kB * kSQ, blk128, 0, stream>>>(Hln, mlp, ln1_g, ln1_b, (float*)d_out, nullptr);

  (void)in_sizes; (void)n_in; (void)out_size; (void)ws_size;
}
